// TransformerBlock_64716567216825
// MI455X (gfx1250) — matrix operand order repaired and checked by reading, not compiled, not run on hardware
//
#include <hip/hip_runtime.h>

// ---------------------------------------------------------------------------
// Transformer block for MI455X (gfx1250): all GEMMs + attention via
// v_wmma_f32_16x16x32_bf16 (bf16 in, f32 accumulate), wave32.
// GEMM staging uses CDNA5 async global->LDS copies (ASYNCcnt) + double buffer.
// ---------------------------------------------------------------------------

#define DEVI __device__ __forceinline__

typedef __bf16 bf16;
typedef __attribute__((ext_vector_type(16))) __bf16 v16bf;
typedef __attribute__((ext_vector_type(8)))  __bf16 v8bf;
typedef __attribute__((ext_vector_type(8)))  float  v8f;
typedef __attribute__((ext_vector_type(4)))  int    v4i;

typedef __attribute__((address_space(1))) v4i* gptr4;   // global int4*
typedef __attribute__((address_space(3))) v4i* lptr4;   // LDS int4*

static constexpr int D_MODEL = 1024;
static constexpr int N_HEADS = 16;
static constexpr int D_FF    = 4096;
static constexpr int BB      = 2;
static constexpr int SS      = 2048;
static constexpr int MROWS   = BB * SS;  // 4096

// ---- CDNA5 async global->LDS support (guarded so compile can't break) ----
#if defined(__has_builtin)
#  if __has_builtin(__builtin_amdgcn_global_load_async_to_lds_b128)
#    define HAVE_ASYNC_LDS 1
#  else
#    define HAVE_ASYNC_LDS 0
#  endif
#  if __has_builtin(__builtin_amdgcn_s_wait_asynccnt)
#    define HAVE_WAIT_ASYNC_BUILTIN 1
#  else
#    define HAVE_WAIT_ASYNC_BUILTIN 0
#  endif
#else
#  define HAVE_ASYNC_LDS 0
#  define HAVE_WAIT_ASYNC_BUILTIN 0
#endif

// Copy 16 bytes per lane from global to LDS (async, tracked by ASYNCcnt).
DEVI void async_copy_b128(const bf16* g, bf16* l) {
#if HAVE_ASYNC_LDS
  __builtin_amdgcn_global_load_async_to_lds_b128((gptr4)(void*)(bf16*)g,
                                                 (lptr4)(void*)l, 0, 0);
#else
  *(v8bf*)l = *(const v8bf*)g;
#endif
}

DEVI void wait_async_copies() {
#if HAVE_ASYNC_LDS
#  if HAVE_WAIT_ASYNC_BUILTIN
  __builtin_amdgcn_s_wait_asynccnt(0);
#  else
  asm volatile("s_wait_asynccnt 0x0" ::: "memory");
#  endif
#endif
}

DEVI v8f zero8() { v8f z = {0.f,0.f,0.f,0.f,0.f,0.f,0.f,0.f}; return z; }

DEVI v16bf combine16(v8bf lo, v8bf hi8) {
  return __builtin_shufflevector(lo, hi8, 0,1,2,3,4,5,6,7,8,9,10,11,12,13,14,15);
}

// A-operand fragment (16x32 bf16, M=row per lane%16).
// Lanes 0-15: K = {0..7, 16..23}; lanes 16-31: K = {8..15, 24..31}.
DEVI v16bf load_frag_a(const bf16* rowPtr, int hi) {
  v8bf lo = *(const v8bf*)(rowPtr + (hi ? 8 : 0));
  v8bf h8 = *(const v8bf*)(rowPtr + 16 + (hi ? 8 : 0));
  return combine16(lo, h8);
}

// B-operand fragment (32x16 bf16, N=col per lane%16), colPtr = &B^T[col][k0].
// Lanes 0-15: K = {0..7, 16..23}; lanes 16-31: K = {8..15, 24..31}.
DEVI v16bf load_frag_b(const bf16* colPtr, int hi) {
  v8bf lo = *(const v8bf*)(colPtr + (hi ? 8 : 0));
  v8bf h8 = *(const v8bf*)(colPtr + 16 + (hi ? 8 : 0));
  return combine16(lo, h8);
}

DEVI v8f wmma_bf16(v16bf a, v16bf b, v8f c) {
  return __builtin_amdgcn_wmma_f32_16x16x32_bf16(
      /*neg_a=*/false, a, /*neg_b=*/false, b,
      /*c_mod=*/(short)0, c, /*reuse_a=*/false, /*reuse_b=*/false);
}

// ---------------------------------------------------------------------------
// Weight transpose + fp32 -> bf16 convert:  src[K][N] -> dst[N][K]
// ---------------------------------------------------------------------------
__global__ __launch_bounds__(256)
void transpose_cvt_kernel(const float* __restrict__ src, bf16* __restrict__ dst,
                          int K, int N) {
  int idx = blockIdx.x * 256 + threadIdx.x;
  if (idx < K * N) {
    int k = idx / N, n = idx % N;
    dst[(size_t)n * K + k] = (bf16)src[idx];
  }
}

// ---------------------------------------------------------------------------
// LayerNorm (row per block, D=1024, 256 threads) -> bf16 activations
// ---------------------------------------------------------------------------
__global__ __launch_bounds__(256)
void layernorm_bf16_kernel(const float* __restrict__ x, const float* __restrict__ w,
                           const float* __restrict__ bprm, bf16* __restrict__ out,
                           int D) {
  const int row = blockIdx.x;
  const int tid = threadIdx.x;
  const float* xr = x + (size_t)row * D;
  __shared__ float red[8];

  float v[4];
  float s = 0.f;
#pragma unroll
  for (int i = 0; i < 4; ++i) { v[i] = xr[tid + 256 * i]; s += v[i]; }
#pragma unroll
  for (int m = 1; m < 32; m <<= 1) s += __shfl_xor(s, m, 32);
  if ((tid & 31) == 0) red[tid >> 5] = s;
  __syncthreads();
  float tot = 0.f;
#pragma unroll
  for (int i = 0; i < 8; ++i) tot += red[i];
  const float mu = tot * (1.0f / (float)D);

  float vs = 0.f;
#pragma unroll
  for (int i = 0; i < 4; ++i) { float d = v[i] - mu; vs += d * d; }
#pragma unroll
  for (int m = 1; m < 32; m <<= 1) vs += __shfl_xor(vs, m, 32);
  __syncthreads();
  if ((tid & 31) == 0) red[tid >> 5] = vs;
  __syncthreads();
  float tot2 = 0.f;
#pragma unroll
  for (int i = 0; i < 8; ++i) tot2 += red[i];
  const float inv = rsqrtf(tot2 * (1.0f / (float)D) + 1e-5f);

#pragma unroll
  for (int i = 0; i < 4; ++i) {
    int c = tid + 256 * i;
    out[(size_t)row * D + c] = (bf16)((v[i] - mu) * inv * w[c] + bprm[c]);
  }
}

// ---------------------------------------------------------------------------
// Tiled WMMA GEMM: C[M,N] = A[M,K](bf16) * Bt[N,K]^T(bf16) + bias (+res) (relu)
// Block: 256 thr = 8 waves. Block tile 128x128, wave tile 32x64.
// Double-buffered LDS, filled with async global->LDS copies (ASYNCcnt).
// out_mode: 0 = fp32 row-major, 1 = bf16 row-major, 2 = bf16 V^T per head.
// ---------------------------------------------------------------------------
__global__ __launch_bounds__(256)
void gemm_bf16_kernel(const bf16* __restrict__ A, const bf16* __restrict__ Bt,
                      const float* __restrict__ bias, const float* __restrict__ residual,
                      void* __restrict__ Cout, int M, int N, int K,
                      int out_mode, int relu, int Svt, int Hvt) {
  __shared__ bf16 As[2][128][40];  // 32 K + 8 pad, double buffered
  __shared__ bf16 Bs[2][128][40];

  const int tid  = threadIdx.x;
  const int lane = tid & 31;
  const int wave = tid >> 5;
  const int hi   = lane >> 4;
  const int l16  = lane & 15;
  const int m0   = blockIdx.y * 128;
  const int n0   = blockIdx.x * 128;
  const int wr   = wave >> 1;  // 0..3 -> 32-row band
  const int wc   = wave & 1;   // 0..1 -> 64-col band

  const int srow  = tid >> 1;        // 0..127 staging row
  const int shalf = (tid & 1) * 16;  // 0 or 16 (k offset)

  const bf16* aBase = A  + (size_t)(m0 + srow) * K + shalf;
  const bf16* bBase = Bt + (size_t)(n0 + srow) * K + shalf;

  auto stage = [&](int bufIdx, int k0) {
    async_copy_b128(aBase + k0,     &As[bufIdx][srow][shalf]);
    async_copy_b128(aBase + k0 + 8, &As[bufIdx][srow][shalf + 8]);
    async_copy_b128(bBase + k0,     &Bs[bufIdx][srow][shalf]);
    async_copy_b128(bBase + k0 + 8, &Bs[bufIdx][srow][shalf + 8]);
  };

  v8f acc[2][4];
#pragma unroll
  for (int i = 0; i < 2; ++i)
#pragma unroll
    for (int j = 0; j < 4; ++j) acc[i][j] = zero8();

  // Prologue: fill buffer 0
  stage(0, 0);
  wait_async_copies();
  __syncthreads();

  int buf = 0;
  for (int k0 = 0; k0 < K; k0 += 32) {
    // Kick off next tile into the other buffer while we compute this one.
    if (k0 + 32 < K) {
      stage(buf ^ 1, k0 + 32);
      if (k0 + 64 < K) {  // prefetch tile after next into L2
        __builtin_prefetch(aBase + k0 + 64, 0, 1);
        __builtin_prefetch(bBase + k0 + 64, 0, 1);
      }
    }

    v16bf af[2], bfr[4];
#pragma unroll
    for (int i = 0; i < 2; ++i)
      af[i] = load_frag_a(&As[buf][wr * 32 + i * 16 + l16][0], hi);
#pragma unroll
    for (int j = 0; j < 4; ++j)
      bfr[j] = load_frag_b(&Bs[buf][wc * 64 + j * 16 + l16][0], hi);
#pragma unroll
    for (int i = 0; i < 2; ++i)
#pragma unroll
      for (int j = 0; j < 4; ++j)
        acc[i][j] = wmma_bf16(af[i], bfr[j], acc[i][j]);

    // Next buffer filled + everyone done reading current buffer.
    wait_async_copies();
    __syncthreads();
    buf ^= 1;
  }

  // Epilogue. C layout: VGPR r -> row (r + 8*hi), lane%16 -> column.
#pragma unroll
  for (int i = 0; i < 2; ++i) {
#pragma unroll
    for (int j = 0; j < 4; ++j) {
      const int col = n0 + wc * 64 + j * 16 + l16;
      const float bval = bias ? bias[col] : 0.f;
#pragma unroll
      for (int r = 0; r < 8; ++r) {
        const int row = m0 + wr * 32 + i * 16 + hi * 8 + r;
        float v = acc[i][j][r] + bval;
        if (residual) v += residual[(size_t)row * N + col];
        if (relu) v = fmaxf(v, 0.f);
        if (out_mode == 0) {
          ((float*)Cout)[(size_t)row * N + col] = v;
        } else if (out_mode == 1) {
          ((bf16*)Cout)[(size_t)row * N + col] = (bf16)v;
        } else {
          // V^T per head: [B, H, Dk, S]
          const int b = row / Svt, s = row % Svt;
          const int hd = col >> 6, dk = col & 63;
          ((bf16*)Cout)[((((size_t)b * Hvt + hd) * 64) + dk) * Svt + s] = (bf16)v;
        }
      }
    }
  }
}

// ---------------------------------------------------------------------------
// Flash attention: block = 128 thr (4 waves), wave owns 16 query rows.
// Q,K bf16 [B,S,D]; Vt bf16 [B,H,Dk,S]; ctx bf16 [B,S,D].
// ---------------------------------------------------------------------------
__global__ __launch_bounds__(128)
void attn_kernel(const bf16* __restrict__ Q, const bf16* __restrict__ Km,
                 const bf16* __restrict__ Vt, bf16* __restrict__ ctx,
                 int Bn, int Hn, int Sn, int Dm) {
  __shared__ bf16 plds[4][16][40];  // per-wave 16x32 P tile (padded)

  const int lane = threadIdx.x & 31;
  const int wave = threadIdx.x >> 5;
  const int hi   = lane >> 4;
  const int l16  = lane & 15;
  const int b    = blockIdx.z;
  const int h    = blockIdx.y;
  const int q0   = blockIdx.x * 64 + wave * 16;

  // Q fragments for dk 0..31 and 32..63 (loaded once, A layout)
  const bf16* qrow = Q + ((size_t)(b * Sn + q0 + l16)) * Dm + h * 64;
  v16bf qf[2];
  qf[0] = load_frag_a(qrow, hi);
  qf[1] = load_frag_a(qrow + 32, hi);

  v8f o[4];
#pragma unroll
  for (int f = 0; f < 4; ++f) o[f] = zero8();
  float mrow[8], lrow[8];
#pragma unroll
  for (int r = 0; r < 8; ++r) { mrow[r] = -1e30f; lrow[r] = 0.f; }

  const float scale = 0.125f;  // 1/sqrt(64)

  for (int key0 = 0; key0 < Sn; key0 += 32) {
    // --- scores S = Q * K^T for a 16x32 key chunk (4 WMMAs) ---
    v8f sc[2];
    sc[0] = zero8(); sc[1] = zero8();
#pragma unroll
    for (int j = 0; j < 2; ++j) {
      const bf16* krow = Km + ((size_t)(b * Sn + key0 + j * 16 + l16)) * Dm + h * 64;
      v16bf kf0 = load_frag_b(krow, hi);       // dk 0..31
      v16bf kf1 = load_frag_b(krow + 32, hi);  // dk 32..63
      sc[j] = wmma_bf16(qf[0], kf0, sc[j]);
      sc[j] = wmma_bf16(qf[1], kf1, sc[j]);
    }

    // --- online softmax (per row r+8*hi; 16 cols live in 16-lane half) ---
    float alpha[8];
#pragma unroll
    for (int r = 0; r < 8; ++r) {
      float v0 = sc[0][r] * scale;
      float v1 = sc[1][r] * scale;
      float mx = fmaxf(v0, v1);
#pragma unroll
      for (int m = 1; m < 16; m <<= 1) mx = fmaxf(mx, __shfl_xor(mx, m, 32));
      const float mnew = fmaxf(mrow[r], mx);
      alpha[r] = __expf(mrow[r] - mnew);
      mrow[r] = mnew;
      const float p0 = __expf(v0 - mnew);
      const float p1 = __expf(v1 - mnew);
      sc[0][r] = p0; sc[1][r] = p1;
      float ps = p0 + p1;
#pragma unroll
      for (int m = 1; m < 16; m <<= 1) ps += __shfl_xor(ps, m, 32);
      lrow[r] = lrow[r] * alpha[r] + ps;
    }

    // --- re-layout P (C frag -> A frag) through per-wave LDS ---
#pragma unroll
    for (int j = 0; j < 2; ++j)
#pragma unroll
      for (int r = 0; r < 8; ++r)
        plds[wave][hi * 8 + r][j * 16 + l16] = (bf16)sc[j][r];

#pragma unroll
    for (int f = 0; f < 4; ++f)
#pragma unroll
      for (int r = 0; r < 8; ++r) o[f][r] *= alpha[r];

    // LDS ops are in-order within a wave: safe to read our own tile back.
    v16bf pf = load_frag_a(&plds[wave][l16][0], hi);

    // --- O += P * V  (V^T rows are contiguous in key dim) ---
#pragma unroll
    for (int f = 0; f < 4; ++f) {
      const bf16* vrow =
          Vt + (((size_t)(b * Hn + h)) * 64 + f * 16 + l16) * Sn + key0;
      v16bf vf = load_frag_b(vrow, hi);
      o[f] = wmma_bf16(pf, vf, o[f]);
    }
  }

  // --- normalize and store ctx ---
#pragma unroll
  for (int f = 0; f < 4; ++f) {
#pragma unroll
    for (int r = 0; r < 8; ++r) {
      const float v = o[f][r] / lrow[r];
      const int row = q0 + hi * 8 + r;
      ctx[((size_t)(b * Sn + row)) * Dm + h * 64 + f * 16 + l16] = (bf16)v;
    }
  }
}

// ---------------------------------------------------------------------------
// Host-side orchestration
// ---------------------------------------------------------------------------
extern "C" void kernel_launch(void* const* d_in, const int* in_sizes, int n_in,
                              void* d_out, int out_size, void* d_ws, size_t ws_size,
                              hipStream_t stream) {
  (void)in_sizes; (void)n_in; (void)out_size; (void)ws_size;

  const float* x    = (const float*)d_in[0];
  const float* ln1w = (const float*)d_in[1];
  const float* ln1b = (const float*)d_in[2];
  const float* Wq   = (const float*)d_in[3];
  const float* bq   = (const float*)d_in[4];
  const float* Wk   = (const float*)d_in[5];
  const float* bk   = (const float*)d_in[6];
  const float* Wv   = (const float*)d_in[7];
  const float* bv   = (const float*)d_in[8];
  const float* Wo   = (const float*)d_in[9];
  const float* bo   = (const float*)d_in[10];
  const float* ln2w = (const float*)d_in[11];
  const float* ln2b = (const float*)d_in[12];
  const float* W1   = (const float*)d_in[13];
  const float* b1   = (const float*)d_in[14];
  const float* W2   = (const float*)d_in[15];
  const float* b2   = (const float*)d_in[16];
  float* out = (float*)d_out;

  // Workspace carve-up (~112 MB total)
  char* ws = (char*)d_ws;
  size_t off = 0;
  auto carve = [&](size_t bytes) -> void* {
    void* p = ws + off;
    off = (off + bytes + 255) & ~(size_t)255;
    return p;
  };
  bf16* WqT = (bf16*)carve((size_t)D_MODEL * D_MODEL * 2);
  bf16* WkT = (bf16*)carve((size_t)D_MODEL * D_MODEL * 2);
  bf16* WvT = (bf16*)carve((size_t)D_MODEL * D_MODEL * 2);
  bf16* WoT = (bf16*)carve((size_t)D_MODEL * D_MODEL * 2);
  bf16* W1T = (bf16*)carve((size_t)D_MODEL * D_FF * 2);
  bf16* W2T = (bf16*)carve((size_t)D_FF * D_MODEL * 2);
  bf16* hb  = (bf16*)carve((size_t)MROWS * D_MODEL * 2);
  bf16* qb  = (bf16*)carve((size_t)MROWS * D_MODEL * 2);
  bf16* kb  = (bf16*)carve((size_t)MROWS * D_MODEL * 2);
  bf16* vtb = (bf16*)carve((size_t)MROWS * D_MODEL * 2);
  bf16* cxb = (bf16*)carve((size_t)MROWS * D_MODEL * 2);
  float* x2 = (float*)carve((size_t)MROWS * D_MODEL * 4);
  bf16* ff1 = (bf16*)carve((size_t)MROWS * D_FF * 2);

  // 1) Weight transpose + bf16 convert
  {
    int n = D_MODEL * D_MODEL;
    dim3 g((n + 255) / 256);
    transpose_cvt_kernel<<<g, 256, 0, stream>>>(Wq, WqT, D_MODEL, D_MODEL);
    transpose_cvt_kernel<<<g, 256, 0, stream>>>(Wk, WkT, D_MODEL, D_MODEL);
    transpose_cvt_kernel<<<g, 256, 0, stream>>>(Wv, WvT, D_MODEL, D_MODEL);
    transpose_cvt_kernel<<<g, 256, 0, stream>>>(Wo, WoT, D_MODEL, D_MODEL);
    int n2 = D_MODEL * D_FF;
    dim3 g2((n2 + 255) / 256);
    transpose_cvt_kernel<<<g2, 256, 0, stream>>>(W1, W1T, D_MODEL, D_FF);
    transpose_cvt_kernel<<<g2, 256, 0, stream>>>(W2, W2T, D_FF, D_MODEL);
  }

  // 2) LN1 -> h (bf16)
  layernorm_bf16_kernel<<<dim3(MROWS), 256, 0, stream>>>(x, ln1w, ln1b, hb, D_MODEL);

  // 3) QKV projections (V written transposed per head)
  {
    dim3 g(D_MODEL / 128, MROWS / 128);
    gemm_bf16_kernel<<<g, 256, 0, stream>>>(hb, WqT, bq, nullptr, qb,
        MROWS, D_MODEL, D_MODEL, /*mode=*/1, /*relu=*/0, 0, 0);
    gemm_bf16_kernel<<<g, 256, 0, stream>>>(hb, WkT, bk, nullptr, kb,
        MROWS, D_MODEL, D_MODEL, 1, 0, 0, 0);
    gemm_bf16_kernel<<<g, 256, 0, stream>>>(hb, WvT, bv, nullptr, vtb,
        MROWS, D_MODEL, D_MODEL, /*mode=*/2, 0, SS, N_HEADS);
  }

  // 4) Attention -> ctx (bf16)
  attn_kernel<<<dim3(SS / 64, N_HEADS, BB), 128, 0, stream>>>(
      qb, kb, vtb, cxb, BB, N_HEADS, SS, D_MODEL);

  // 5) Output projection + residual -> x2 (fp32)
  gemm_bf16_kernel<<<dim3(D_MODEL / 128, MROWS / 128), 256, 0, stream>>>(
      cxb, WoT, bo, x, x2, MROWS, D_MODEL, D_MODEL, /*mode=*/0, 0, 0, 0);

  // 6) LN2 -> h (bf16, reuse buffer)
  layernorm_bf16_kernel<<<dim3(MROWS), 256, 0, stream>>>(x2, ln2w, ln2b, hb, D_MODEL);

  // 7) FFN up + ReLU -> ff1 (bf16)
  gemm_bf16_kernel<<<dim3(D_FF / 128, MROWS / 128), 256, 0, stream>>>(
      hb, W1T, b1, nullptr, ff1, MROWS, D_FF, D_MODEL, /*mode=*/1, /*relu=*/1, 0, 0);

  // 8) FFN down + residual -> out (fp32)
  gemm_bf16_kernel<<<dim3(D_MODEL / 128, MROWS / 128), 256, 0, stream>>>(
      ff1, W2T, b2, x2, out, MROWS, D_MODEL, D_FF, /*mode=*/0, 0, 0, 0);
}
